// Qwen2HareMoE_56667798504235
// MI455X (gfx1250) — compile-verified, matrix operand order checked by reading, not run on hardware
//
#include <hip/hip_runtime.h>

// ---------------- problem sizes (compile-time) ----------------
#define T_TOK 4096
#define DIM   2048
#define FFN_  1408
#define NEXP  16
#define SFFN  5632
#define TOPK  4

typedef __attribute__((ext_vector_type(8)))  float          v8f;
typedef __attribute__((ext_vector_type(16))) __bf16         v16bf;
typedef __attribute__((ext_vector_type(8)))  unsigned short u16x8;
typedef __attribute__((ext_vector_type(4)))  unsigned int   u32x4;
typedef __attribute__((ext_vector_type(8)))  unsigned int   u32x8;

// native fp32 -> bf16 convert (v_cvt*_bf16_f32), bit pattern out
__device__ __forceinline__ unsigned short f2bf(float f) {
    union { __bf16 h; unsigned short u; } cv;
    cv.h = (__bf16)f;
    return cv.u;
}

// build a 16x16 WMMA B/A fragment from 16 consecutive fp32 in LDS
__device__ __forceinline__ v16bf frag_from_f32(const float* src) {
    v16bf t;
#pragma unroll
    for (int q = 0; q < 16; ++q) t[q] = (__bf16)src[q];
    return t;
}

// ---- CDNA5 async global->LDS copy: 32 bytes/lane as 2x b128 (ASYNCcnt) ----
__device__ __forceinline__ void async_copy32(const void* g, unsigned lds_off) {
    unsigned long long ga = (unsigned long long)g;
    asm volatile("global_load_async_to_lds_b128 %0, %1, off"
                 :: "v"(lds_off), "v"(ga) : "memory");
    asm volatile("global_load_async_to_lds_b128 %0, %1, off offset:16"
                 :: "v"(lds_off), "v"(ga) : "memory");
}
__device__ __forceinline__ void wait_async_tile(int pending_tiles) {
    if (pending_tiles) asm volatile("s_wait_asynccnt 0x2" ::: "memory");
    else               asm volatile("s_wait_asynccnt 0x0" ::: "memory");
}

// ---- CDNA5 Tensor Data Mover: 2-D fp32 tile [tile1 rows x tile0 cols],
//      row stride stride0 (elements), global -> LDS (TENSORcnt) ----
__device__ __forceinline__ void tdm_load_2d_f32(
        const void* gaddr, unsigned lds_addr,
        unsigned tile0, unsigned tile1,
        unsigned long long stride0, unsigned td0, unsigned td1) {
    unsigned long long ga = (unsigned long long)gaddr;
    u32x4 g0;
    g0[0] = 1u;                                            // count=1, user mode
    g0[1] = lds_addr;                                      // LDS byte address
    g0[2] = (unsigned)(ga & 0xFFFFFFFFu);                  // global_addr[31:0]
    g0[3] = (unsigned)((ga >> 32) & 0x1FFFFFFu) | (2u << 30); // [56:32] | type=2
    u32x8 g1;
    g1[0] = 0x00020000u;                                   // data_size=4B, mask=0
    g1[1] = (td0 & 0xFFFFu) << 16;                         // tensor_dim0[15:0]
    g1[2] = (td0 >> 16) | ((td1 & 0xFFFFu) << 16);         // td0[31:16] | td1[15:0]
    g1[3] = (td1 >> 16) | (tile0 << 16);                   // td1[31:16] | tile_dim0
    g1[4] = tile1;                                         // tile_dim1 (tile_dim2=0)
    g1[5] = (unsigned)(stride0 & 0xFFFFFFFFu);             // dim0_stride[31:0]
    g1[6] = (unsigned)((stride0 >> 32) & 0xFFFFu);         // dim0_stride[47:32]
    g1[7] = 0u;
    asm volatile("tensor_load_to_lds %0, %1" :: "s"(g0), "s"(g1) : "memory");
}

// =====================================================================
// Router / dispatch kernels
// =====================================================================
__global__ void moe_init_kernel(int* counts, int* cursor) {
    int i = threadIdx.x;
    if (i < NEXP) { counts[i] = 0; cursor[i] = 0; }
}

__global__ __launch_bounds__(256) void moe_cvt_x_kernel(
        const float* __restrict__ x, unsigned short* __restrict__ xb) {
    size_t i = ((size_t)blockIdx.x * 256 + threadIdx.x) * 8;
    u16x8 pk;
#pragma unroll
    for (int k = 0; k < 8; ++k) pk[k] = f2bf(x[i + k]);
    *(u16x8*)(xb + i) = pk;
}

__global__ __launch_bounds__(256) void moe_logits_kernel(
        const float* __restrict__ x, const float* __restrict__ gate_w,
        const float* __restrict__ seg_w, float* __restrict__ logits) {
    int idx = blockIdx.x * 256 + threadIdx.x;
    if (idx >= T_TOK * 17) return;
    int t = idx / 17, e = idx % 17;
    const float* wr = (e < NEXP) ? (gate_w + (size_t)e * DIM) : seg_w;
    const float* xr = x + (size_t)t * DIM;
    float acc = 0.f;
    for (int d = 0; d < DIM; d += 4) {
        acc += xr[d] * wr[d] + xr[d + 1] * wr[d + 1]
             + xr[d + 2] * wr[d + 2] + xr[d + 3] * wr[d + 3];
    }
    logits[idx] = acc;
}

__global__ __launch_bounds__(256) void moe_route_kernel(
        const float* __restrict__ logits, int* __restrict__ ids,
        float* __restrict__ wts, float* __restrict__ gate,
        int* __restrict__ counts) {
    int t = blockIdx.x * 256 + threadIdx.x;
    if (t >= T_TOK) return;
    float p[NEXP];
    float mx = -3.4e38f;
#pragma unroll
    for (int e = 0; e < NEXP; ++e) { p[e] = logits[t * 17 + e]; mx = fmaxf(mx, p[e]); }
    float sum = 0.f;
#pragma unroll
    for (int e = 0; e < NEXP; ++e) { p[e] = __expf(p[e] - mx); sum += p[e]; }
    float inv = 1.0f / sum;
#pragma unroll
    for (int e = 0; e < NEXP; ++e) p[e] *= inv;

    float wsum = 0.f;
#pragma unroll
    for (int k = 0; k < TOPK; ++k) {
        int   bi = 0;
        float bv = -1.f;
#pragma unroll
        for (int e = 0; e < NEXP; ++e)
            if (p[e] > bv) { bv = p[e]; bi = e; }
        ids[t * TOPK + k] = bi;
        wts[t * TOPK + k] = bv;
        wsum += bv;
        p[bi] = -1.f;
        atomicAdd(&counts[bi], 1);
    }
    float winv = 1.0f / wsum;
#pragma unroll
    for (int k = 0; k < TOPK; ++k) wts[t * TOPK + k] *= winv;
    gate[t] = 1.0f / (1.0f + __expf(-logits[t * 17 + NEXP]));
}

__global__ void moe_scan_kernel(const int* counts, int* offs) {
    if (threadIdx.x == 0) {
        int acc = 0;
        offs[0] = 0;
        for (int e = 0; e < NEXP; ++e) { acc += counts[e]; offs[e + 1] = acc; }
    }
}

__global__ __launch_bounds__(256) void moe_build_kernel(
        const int* __restrict__ ids, const float* __restrict__ wts,
        const int* __restrict__ offs, int* __restrict__ cursor,
        int* __restrict__ perm, float* __restrict__ pwgt) {
    int idx = blockIdx.x * 256 + threadIdx.x;
    if (idx >= T_TOK * TOPK) return;
    int t = idx / TOPK;
    int e = ids[idx];
    int pos = atomicAdd(&cursor[e], 1);
    int slot = offs[e] + pos;
    perm[slot] = t;
    pwgt[slot] = wts[idx];
}

// =====================================================================
// SwiGLU up/gate GEMM: H = bf16( silu(A@B1^T) * (A@B3^T) )
// Block 128x64xK32, 8 waves (4Mx2N), wave = 2x2 WMMA 16x16x32 bf16.
// A (bf16 activations): async global->LDS, double buffered (ASYNCcnt).
// B1/B3 (fp32 weights): Tensor Data Mover 2-D tiles, double buffered
// (TENSORcnt), converted to bf16 at fragment-load time.
// =====================================================================
__global__ __launch_bounds__(256) void moe_swiglu_gemm(
        const unsigned short* __restrict__ A, const float* __restrict__ B1,
        const float* __restrict__ B3, unsigned short* __restrict__ H,
        const int* __restrict__ perm, const int* __restrict__ offs,
        int expert_mode, int N, int K) {
    int e = blockIdx.z;
    int row0 = 0, rows;
    const float *b1 = B1, *b3 = B3;
    if (expert_mode) {
        row0 = offs[e];
        rows = offs[e + 1] - row0;
        size_t wsz = (size_t)N * (size_t)K;
        b1 = B1 + (size_t)e * wsz;
        b3 = B3 + (size_t)e * wsz;
    } else {
        rows = T_TOK;
    }
    int mtile = blockIdx.y;
    if (mtile * 128 >= rows) return;            // uniform exit
    int ntile = blockIdx.x;
    int rows_m = rows - mtile * 128; if (rows_m > 128) rows_m = 128;

    __shared__ __align__(32) unsigned short sA[2][128 * 32];
    __shared__ __align__(32) float sB1f[2][64 * 32];
    __shared__ __align__(32) float sB3f[2][64 * 32];

    int tid  = threadIdx.x;
    int lane = tid & 31, wave = tid >> 5;
    int wm = wave & 3, wn = wave >> 2;
    int ln = lane & 15, lh = lane >> 4;

    // A staging addresses (2 threads/row, 32B each)
    int ar = tid >> 1, aseg = tid & 1;
    int tokA = 0;
    if (ar < rows_m)
        tokA = expert_mode ? perm[row0 + mtile * 128 + ar] : (mtile * 128 + ar);
    const unsigned short* aSrc = A + (size_t)tokA * (size_t)K + aseg * 16;
    unsigned aoff[2] = { (unsigned)(size_t)&sA[0][ar * 32 + aseg * 16],
                         (unsigned)(size_t)&sA[1][ar * 32 + aseg * 16] };
    unsigned b1off[2] = { (unsigned)(size_t)&sB1f[0][0],
                          (unsigned)(size_t)&sB1f[1][0] };
    unsigned b3off[2] = { (unsigned)(size_t)&sB3f[0][0],
                          (unsigned)(size_t)&sB3f[1][0] };

    const float* bTile1 = b1 + (size_t)(ntile * 64) * (size_t)K;  // 64 rows x K
    const float* bTile3 = b3 + (size_t)(ntile * 64) * (size_t)K;

    v8f acc1[2][2], acc3[2][2];
#pragma unroll
    for (int i = 0; i < 2; ++i)
#pragma unroll
        for (int j = 0; j < 2; ++j) { acc1[i][j] = (v8f){}; acc3[i][j] = (v8f){}; }

    // prologue: tile 0 in flight (A async + both B TDM)
    async_copy32(aSrc, aoff[0]);
    if (wave == 0) {
        tdm_load_2d_f32(bTile1, b1off[0], 32u, 64u, (unsigned long long)K,
                        (unsigned)K, 64u);
        tdm_load_2d_f32(bTile3, b3off[0], 32u, 64u, (unsigned long long)K,
                        (unsigned)K, 64u);
    }

    int cb = 0;
    for (int k0 = 0; k0 < K; k0 += 32) {
        int nb = cb ^ 1;
        int hasNext = (k0 + 32) < K;
        if (hasNext) {
            async_copy32(aSrc + k0 + 32, aoff[nb]);
            if (wave == 0) {
                tdm_load_2d_f32(bTile1 + k0 + 32, b1off[nb], 32u, 64u,
                                (unsigned long long)K, (unsigned)K, 64u);
                tdm_load_2d_f32(bTile3 + k0 + 32, b3off[nb], 32u, 64u,
                                (unsigned long long)K, (unsigned)K, 64u);
            }
        }
        if (wave == 0) {
            if (hasNext) __builtin_amdgcn_s_wait_tensorcnt(2);
            else         __builtin_amdgcn_s_wait_tensorcnt(0);
        }
        wait_async_tile(hasNext);
        __syncthreads();

        v16bf bf1[2], bf3[2];
#pragma unroll
        for (int j = 0; j < 2; ++j) {
            int boff = (wn * 32 + j * 16 + ln) * 32 + lh * 16;
            bf1[j] = frag_from_f32(&sB1f[cb][boff]);
            bf3[j] = frag_from_f32(&sB3f[cb][boff]);
        }
#pragma unroll
        for (int i = 0; i < 2; ++i) {
            v16bf af = *(const v16bf*)(&sA[cb][(wm * 32 + i * 16 + ln) * 32 + lh * 16]);
#pragma unroll
            for (int j = 0; j < 2; ++j) {
                acc1[i][j] = __builtin_amdgcn_wmma_f32_16x16x32_bf16(
                    false, af, false, bf1[j], (short)0, acc1[i][j], false, false);
                acc3[i][j] = __builtin_amdgcn_wmma_f32_16x16x32_bf16(
                    false, af, false, bf3[j], (short)0, acc3[i][j], false, false);
            }
        }
        __syncthreads();
        cb = nb;
    }

#pragma unroll
    for (int i = 0; i < 2; ++i)
#pragma unroll
        for (int j = 0; j < 2; ++j)
#pragma unroll
            for (int r = 0; r < 8; ++r) {
                int lrow = wm * 32 + i * 16 + r + 8 * lh;
                int grow = mtile * 128 + lrow;
                if (grow < rows) {
                    int gcol = ntile * 64 + wn * 32 + j * 16 + ln;
                    float g = acc1[i][j][r];
                    float u = acc3[i][j][r];
                    float hv = (g / (1.0f + __expf(-g))) * u;
                    H[(size_t)(row0 + grow) * (size_t)N + gcol] = f2bf(hv);
                }
            }
}

// =====================================================================
// Down GEMM: y = A@B^T; expert: atomic scatter-add w*y; shared: gate*y.
// A: bf16 via async global->LDS. B: fp32 weight tile via TDM.
// =====================================================================
__global__ __launch_bounds__(256) void moe_down_gemm(
        const unsigned short* __restrict__ A, const float* __restrict__ B,
        float* __restrict__ out, const int* __restrict__ perm,
        const float* __restrict__ scale, const int* __restrict__ offs,
        int expert_mode, int N, int K) {
    int e = blockIdx.z;
    int row0 = 0, rows;
    const float* b = B;
    if (expert_mode) {
        row0 = offs[e];
        rows = offs[e + 1] - row0;
        b = B + (size_t)e * (size_t)N * (size_t)K;
    } else {
        rows = T_TOK;
    }
    int mtile = blockIdx.y;
    if (mtile * 128 >= rows) return;
    int ntile = blockIdx.x;
    int rows_m = rows - mtile * 128; if (rows_m > 128) rows_m = 128;

    __shared__ __align__(32) unsigned short sA[2][128 * 32];
    __shared__ __align__(32) float sBf[2][64 * 32];

    int tid  = threadIdx.x;
    int lane = tid & 31, wave = tid >> 5;
    int wm = wave & 3, wn = wave >> 2;
    int ln = lane & 15, lh = lane >> 4;

    int ar = tid >> 1, aseg = tid & 1;
    int arow = row0 + ((ar < rows_m) ? (mtile * 128 + ar) : 0);
    const unsigned short* aSrc = A + (size_t)arow * (size_t)K + aseg * 16;
    unsigned aoff[2] = { (unsigned)(size_t)&sA[0][ar * 32 + aseg * 16],
                         (unsigned)(size_t)&sA[1][ar * 32 + aseg * 16] };
    unsigned boff[2] = { (unsigned)(size_t)&sBf[0][0],
                         (unsigned)(size_t)&sBf[1][0] };

    const float* bTile = b + (size_t)(ntile * 64) * (size_t)K;  // 64 rows x K

    v8f acc[2][2];
#pragma unroll
    for (int i = 0; i < 2; ++i)
#pragma unroll
        for (int j = 0; j < 2; ++j) acc[i][j] = (v8f){};

    async_copy32(aSrc, aoff[0]);
    if (wave == 0)
        tdm_load_2d_f32(bTile, boff[0], 32u, 64u, (unsigned long long)K,
                        (unsigned)K, 64u);

    int cb = 0;
    for (int k0 = 0; k0 < K; k0 += 32) {
        int nb = cb ^ 1;
        int hasNext = (k0 + 32) < K;
        if (hasNext) {
            async_copy32(aSrc + k0 + 32, aoff[nb]);
            if (wave == 0)
                tdm_load_2d_f32(bTile + k0 + 32, boff[nb], 32u, 64u,
                                (unsigned long long)K, (unsigned)K, 64u);
        }
        if (wave == 0) {
            if (hasNext) __builtin_amdgcn_s_wait_tensorcnt(1);
            else         __builtin_amdgcn_s_wait_tensorcnt(0);
        }
        wait_async_tile(hasNext);
        __syncthreads();

        v16bf bf[2];
#pragma unroll
        for (int j = 0; j < 2; ++j)
            bf[j] = frag_from_f32(&sBf[cb][(wn * 32 + j * 16 + ln) * 32 + lh * 16]);
#pragma unroll
        for (int i = 0; i < 2; ++i) {
            v16bf af = *(const v16bf*)(&sA[cb][(wm * 32 + i * 16 + ln) * 32 + lh * 16]);
#pragma unroll
            for (int j = 0; j < 2; ++j)
                acc[i][j] = __builtin_amdgcn_wmma_f32_16x16x32_bf16(
                    false, af, false, bf[j], (short)0, acc[i][j], false, false);
        }
        __syncthreads();
        cb = nb;
    }

#pragma unroll
    for (int i = 0; i < 2; ++i)
#pragma unroll
        for (int j = 0; j < 2; ++j)
#pragma unroll
            for (int r = 0; r < 8; ++r) {
                int lrow = wm * 32 + i * 16 + r + 8 * lh;
                int grow = mtile * 128 + lrow;
                if (grow < rows) {
                    int gcol = ntile * 64 + wn * 32 + j * 16 + ln;
                    float v = acc[i][j][r];
                    if (expert_mode) {
                        int cr  = row0 + grow;
                        int tok = perm[cr];
                        atomicAdd(out + (size_t)tok * DIM + gcol, scale[cr] * v);
                    } else {
                        out[(size_t)grow * DIM + gcol] = scale[grow] * v;
                    }
                }
            }
}

// =====================================================================
// Host launcher
// =====================================================================
extern "C" void kernel_launch(void* const* d_in, const int* in_sizes, int n_in,
                              void* d_out, int out_size, void* d_ws, size_t ws_size,
                              hipStream_t stream) {
    const float* x    = (const float*)d_in[0];
    const float* gw   = (const float*)d_in[1];
    const float* w1   = (const float*)d_in[2];
    const float* w2   = (const float*)d_in[3];
    const float* w3   = (const float*)d_in[4];
    const float* sgw  = (const float*)d_in[5];
    const float* suw  = (const float*)d_in[6];
    const float* sdw  = (const float*)d_in[7];
    const float* segw = (const float*)d_in[8];
    float* out = (float*)d_out;

    char* w = (char*)d_ws;
    unsigned short* xb = (unsigned short*)w;  w += (size_t)T_TOK * DIM * 2;
    float* logits = (float*)w;                w += (size_t)T_TOK * 17 * 4;
    float* gate   = (float*)w;                w += (size_t)T_TOK * 4;
    int*   ids    = (int*)w;                  w += (size_t)T_TOK * TOPK * 4;
    float* wts    = (float*)w;                w += (size_t)T_TOK * TOPK * 4;
    int*   counts = (int*)w;                  w += 256;
    int*   offs   = (int*)w;                  w += 256;
    int*   cursor = (int*)w;                  w += 256;
    int*   perm   = (int*)w;                  w += (size_t)T_TOK * TOPK * 4;
    float* pwgt   = (float*)w;                w += (size_t)T_TOK * TOPK * 4;
    unsigned short* hbuf = (unsigned short*)w; w += (size_t)T_TOK * TOPK * FFN_ * 2;
    unsigned short* hs   = (unsigned short*)w; w += (size_t)T_TOK * SFFN * 2;

    moe_init_kernel<<<1, 64, 0, stream>>>(counts, cursor);
    moe_cvt_x_kernel<<<(T_TOK * DIM) / (256 * 8), 256, 0, stream>>>(x, xb);
    moe_logits_kernel<<<(T_TOK * 17 + 255) / 256, 256, 0, stream>>>(x, gw, segw, logits);
    moe_route_kernel<<<T_TOK / 256, 256, 0, stream>>>(logits, ids, wts, gate, counts);
    moe_scan_kernel<<<1, 1, 0, stream>>>(counts, offs);
    moe_build_kernel<<<(T_TOK * TOPK) / 256, 256, 0, stream>>>(ids, wts, offs, cursor, perm, pwgt);

    moe_swiglu_gemm<<<dim3(SFFN / 64, T_TOK / 128, 1), 256, 0, stream>>>(
        xb, sgw, suw, hs, nullptr, nullptr, 0, SFFN, DIM);
    moe_down_gemm<<<dim3(DIM / 64, T_TOK / 128, 1), 256, 0, stream>>>(
        hs, sdw, out, nullptr, gate, nullptr, 0, DIM, SFFN);

    moe_swiglu_gemm<<<dim3(FFN_ / 64, T_TOK / 128, NEXP), 256, 0, stream>>>(
        xb, w1, w3, hbuf, perm, offs, 1, FFN_, DIM);
    moe_down_gemm<<<dim3(DIM / 64, T_TOK / 128, NEXP), 256, 0, stream>>>(
        hbuf, w2, out, perm, pwgt, offs, 1, DIM, FFN_);
}
